// QuantizedLinearPattern_3736621547692
// MI455X (gfx1250) — matrix operand order repaired and checked by reading, not compiled, not run on hardware
//
#include <hip/hip_runtime.h>

// ---------------------------------------------------------------------------
// Quantized linear (int4 codes + per-group scales) as f16 WMMA GEMM.
// y[T, O] = x[T, K] * W^T,  W[o,k] = q[o,k] * scales[o, k/128]
// Block tile 128(M) x 256(N), 8 waves as 2(M) x 4(N), wave tile 64x64.
// K staged 32 per step; ping-pong double-buffered LDS, ONE barrier per step
// so next-tile global staging overlaps the 16 WMMAs of the current tile.
// ---------------------------------------------------------------------------

typedef __attribute__((ext_vector_type(16))) _Float16 v16h;
typedef __attribute__((ext_vector_type(8)))  _Float16 v8h;
typedef __attribute__((ext_vector_type(4)))  _Float16 v4h;
typedef __attribute__((ext_vector_type(8)))  float    v8f;

constexpr int IN_F   = 4096;
constexpr int OUT_F  = 11008;
constexpr int TOKENS = 4096;
constexpr int GROUP  = 128;
constexpr int NGROUP = IN_F / GROUP;    // 32

constexpr int TILE_M = 128;
constexpr int TILE_N = 256;
constexpr int TILE_K = 32;              // one WMMA K-step
constexpr int NSTEPS = IN_F / TILE_K;   // 128
constexpr int A_STR  = TILE_K + 8;      // 40 halves  (80 B row: conflict-free)
constexpr int B_STR  = TILE_N + 8;      // 264 halves (528 B row: 16B-aligned)

__global__ __launch_bounds__(256)
void q4_linear_wmma(const float* __restrict__ x,
                    const int*   __restrict__ qw,
                    const float* __restrict__ scales,
                    float* __restrict__ out)
{
    __shared__ _Float16 ldsA[2][TILE_M * A_STR];   // [m][k] f16 activations
    __shared__ _Float16 ldsB[2][TILE_K * B_STR];   // [k][n] f16 dequant weights

    const int tid  = threadIdx.x;
    const int lane = tid & 31;
    const int wid  = tid >> 5;
    const int wm   = wid & 1;           // 2 waves along M (64 rows each)
    const int wn   = wid >> 1;          // 4 waves along N (64 cols each)

    const int tile_m = blockIdx.y * TILE_M;
    const int tile_n = blockIdx.x * TILE_N;

    // A loader: thread stages 16 contiguous k of one token row (64B load)
    const int a_row = tid >> 1;
    const int a_col = (tid & 1) * 16;
    const float* a_src = x + (size_t)(tile_m + a_row) * IN_F + a_col;

    // B loader: thread stages 8 contiguous k of FOUR adjacent output rows,
    // writing f16 quads into the transposed [k][n] LDS tile (8B stores).
    const int b_n = (tid >> 2) * 4;     // 0..252 (multiple of 4)
    const int b_k = (tid & 3) * 8;      // 0,8,16,24
    const int* b_src = qw + (size_t)(tile_n + b_n) * IN_F + b_k;

    v8f acc[4][4];
    {
        v8f z = {0.f, 0.f, 0.f, 0.f, 0.f, 0.f, 0.f, 0.f};
        #pragma unroll
        for (int i = 0; i < 4; ++i)
            #pragma unroll
            for (int j = 0; j < 4; ++j)
                acc[i][j] = z;
    }

    // Fragment addressing (16-bit A layout: lanes 0-15 hold K0-7 & K16-23,
    // lanes 16-31 hold K8-15 & K24-31; B layout: N = lane&15, same K order as A).
    const int a_frag_row = wm * 64 + (lane & 15);
    const int a_frag_k   = (lane >> 4) * 8;        // 0 or 8
    const int b_frag_n0  = wn * 64;

    // Per-group scales for this thread's four weight rows (f16 is ample:
    // scales are ~1e-3..2e-2 and int4 codes are exact in f16).
    _Float16 hs[4];

    // ---- stage one K-tile (step ks) into LDS buffer p ----
    auto stage = [&](int ks, int p) {
        const int k0 = ks * TILE_K;

        // A: fp32 -> f16
        const float4* xa = reinterpret_cast<const float4*>(a_src + k0);
        float4 f0 = xa[0], f1 = xa[1], f2 = xa[2], f3 = xa[3];
        v8h a0 = {(_Float16)f0.x, (_Float16)f0.y, (_Float16)f0.z, (_Float16)f0.w,
                  (_Float16)f1.x, (_Float16)f1.y, (_Float16)f1.z, (_Float16)f1.w};
        v8h a1 = {(_Float16)f2.x, (_Float16)f2.y, (_Float16)f2.z, (_Float16)f2.w,
                  (_Float16)f3.x, (_Float16)f3.y, (_Float16)f3.z, (_Float16)f3.w};
        *reinterpret_cast<v8h*>(&ldsA[p][a_row * A_STR + a_col])     = a0;
        *reinterpret_cast<v8h*>(&ldsA[p][a_row * A_STR + a_col + 8]) = a1;

        // B: dequant int4 codes -> f16, transposed to [k][n]
        int q[4][8];
        #pragma unroll
        for (int r = 0; r < 4; ++r) {
            const int4* qp =
                reinterpret_cast<const int4*>(b_src + (size_t)r * IN_F + k0);
            int4 lo = qp[0], hi = qp[1];
            q[r][0] = lo.x; q[r][1] = lo.y; q[r][2] = lo.z; q[r][3] = lo.w;
            q[r][4] = hi.x; q[r][5] = hi.y; q[r][6] = hi.z; q[r][7] = hi.w;
        }
        #pragma unroll
        for (int j = 0; j < 8; ++j) {
            v4h pk = {(_Float16)q[0][j] * hs[0],
                      (_Float16)q[1][j] * hs[1],
                      (_Float16)q[2][j] * hs[2],
                      (_Float16)q[3][j] * hs[3]};
            *reinterpret_cast<v4h*>(&ldsB[p][(b_k + j) * B_STR + b_n]) = pk;
        }

        // warm L2 two tiles ahead (global_prefetch_b8)
        if (k0 + 2 * TILE_K < IN_F) {
            __builtin_prefetch(a_src + k0 + 2 * TILE_K, 0, 1);
            __builtin_prefetch(b_src + k0 + 2 * TILE_K, 0, 1);
        }
    };

    // ---- consume LDS buffer p: 16x v_wmma_f32_16x16x32_f16 ----
    auto compute = [&](int p) {
        v16h bfrag[4];
        #pragma unroll
        for (int ni = 0; ni < 4; ++ni) {
            const _Float16* pb = &ldsB[p][((lane >> 4) * 8) * B_STR + b_frag_n0 + ni * 16 + (lane & 15)];
            #pragma unroll
            for (int s = 0; s < 16; ++s)
                bfrag[ni][s] = pb[((s & 7) + (s >> 3) * 16) * B_STR];
        }
        #pragma unroll
        for (int mi = 0; mi < 4; ++mi) {
            const _Float16* pa =
                &ldsA[p][(a_frag_row + mi * 16) * A_STR + a_frag_k];
            v8h lo = *reinterpret_cast<const v8h*>(pa);
            v8h hi = *reinterpret_cast<const v8h*>(pa + 16);
            v16h af = __builtin_shufflevector(lo, hi,
                0, 1, 2, 3, 4, 5, 6, 7, 8, 9, 10, 11, 12, 13, 14, 15);
            #pragma unroll
            for (int ni = 0; ni < 4; ++ni)
                acc[mi][ni] = __builtin_amdgcn_wmma_f32_16x16x32_f16(
                    /*neg_a=*/false, af,
                    /*neg_b=*/false, bfrag[ni],
                    /*c_mod=*/(short)0, acc[mi][ni],
                    /*reuse_a=*/false, /*reuse_b=*/false);
        }
    };

    // ---- prologue: scales for group 0, stage step 0 into buffer 0 ----
    #pragma unroll
    for (int r = 0; r < 4; ++r)
        hs[r] = (_Float16)scales[(size_t)(tile_n + b_n + r) * NGROUP + 0];
    stage(0, 0);
    __syncthreads();

    // ---- main loop: one barrier per K-step, staging runs ahead by one ----
    for (int g = 0; g < NGROUP; ++g) {
        #pragma unroll
        for (int kk = 0; kk < GROUP / TILE_K; ++kk) {
            const int ks = g * (GROUP / TILE_K) + kk;
            const int p  = kk & 1;              // 4*g is even => parity = kk&1

            if (kk == GROUP / TILE_K - 1) {     // next step starts group g+1
                const int gn = g + 1;
                if (gn < NGROUP) {
                    #pragma unroll
                    for (int r = 0; r < 4; ++r)
                        hs[r] = (_Float16)
                            scales[(size_t)(tile_n + b_n + r) * NGROUP + gn];
                }
            }
            if (ks + 1 < NSTEPS)
                stage(ks + 1, p ^ 1);           // overlap with WMMAs below

            compute(p);
            __syncthreads();
        }
    }

    // ---- epilogue: C/D layout — lanes 0-15: M=v, lanes 16-31: M=v+8; N=lane&15
    const int c_m = tile_m + wm * 64 + (lane >> 4) * 8;
    const int c_n = tile_n + wn * 64 + (lane & 15);
    #pragma unroll
    for (int mi = 0; mi < 4; ++mi)
        #pragma unroll
        for (int ni = 0; ni < 4; ++ni)
            #pragma unroll
            for (int v = 0; v < 8; ++v)
                out[(size_t)(c_m + mi * 16 + v) * OUT_F + (c_n + ni * 16)]
                    = acc[mi][ni][v];
}

extern "C" void kernel_launch(void* const* d_in, const int* in_sizes, int n_in,
                              void* d_out, int out_size, void* d_ws, size_t ws_size,
                              hipStream_t stream) {
    const float* x      = (const float*)d_in[0];
    const int*   qw     = (const int*)  d_in[1];
    const float* scales = (const float*)d_in[2];
    float*       out    = (float*)d_out;

    dim3 grid(OUT_F / TILE_N, TOKENS / TILE_M);   // (43, 32)
    dim3 block(256);
    q4_linear_wmma<<<grid, block, 0, stream>>>(x, qw, scales, out);
}
